// VectorQuantizer_60103772340473
// MI455X (gfx1250) — compile-verified, matrix operand order checked by reading, not run on hardware
//
#include <hip/hip_runtime.h>
#include <hip/hip_bf16.h>

typedef __attribute__((ext_vector_type(16))) __bf16 v16bf;
typedef __attribute__((ext_vector_type(8)))  float  v8f;

#define K_CODES   1024
#define C_DIM     64
#define SPATIAL   16384        // 16*32*32 (per (b,c) plane)
#define N_ROWS    65536        // 4 * SPATIAL
#define N_ELEMS   4194304      // N_ROWS * C_DIM

// workspace layout (bytes)
#define WS_FRAG   0            // __bf16[65536]  : swizzled codebook fragments (128 KB)
#define WS_ENORM  131072       // float[1024]    : ||e_k||^2
#define WS_PART   135168       // float[128]     : per-block loss partials
#define SMEM_BYTES 135168      // fragments + enorm staged in LDS

#define MAIN_BLOCKS     128
#define WAVES_PER_BLOCK 8
#define TILE_PAIRS      2      // per wave: 2 pairs = 4 tiles of 16 rows
// 128 blocks * 8 waves * 2 pairs * 32 rows = 65536 rows

__device__ __forceinline__ __bf16 f2bf(float f) { return (__bf16)f; }

// Async global -> LDS copy (gfx1250): one b128 per lane, tracked by ASYNCcnt.
__device__ __forceinline__ void async_g2l_b128(unsigned lds_off, const void* gaddr) {
  asm volatile("global_load_async_to_lds_b128 %0, %1, off"
               :: "v"(lds_off), "v"(gaddr) : "memory");
}
__device__ __forceinline__ void wait_asynccnt0() {
  asm volatile("s_wait_asynccnt 0x0" ::: "memory");
}

// ---------------------------------------------------------------------------
// Kernel 1: build bf16 B-fragments in WMMA layout, ||e||^2, zero partials.
// Fragment f = codeTile*2 + chunk. Per lane l (h=l>>4, r=l&15):
//   column n = r (code = codeTile*16 + r)
//   element i -> k = chunk*32 + i + 8*h + (i<8 ? 0 : 8)
// matching the ISA "16-bit A/B 16x32 / 32x16" VGPR tables.
// ---------------------------------------------------------------------------
__global__ void vq_prep(const float* __restrict__ emb, char* __restrict__ ws) {
  int tid = blockIdx.x * blockDim.x + threadIdx.x;
  __bf16* frag  = (__bf16*)(ws + WS_FRAG);
  float*  enorm = (float*)(ws + WS_ENORM);
  float*  part  = (float*)(ws + WS_PART);
  if (tid < 65536) {
    int f = tid >> 9;            // 512 elements per fragment (32 lanes * 16)
    int l = (tid >> 4) & 31;
    int i = tid & 15;
    int codeTile = f >> 1, chunk = f & 1;
    int h = l >> 4, r = l & 15;
    int code = codeTile * 16 + r;
    int k = chunk * 32 + i + 8 * h + ((i < 8) ? 0 : 8);
    frag[f * 512 + l * 16 + i] = f2bf(emb[code * 64 + k]);
  } else if (tid < 65536 + 1024) {
    int code = tid - 65536;
    const float* e = emb + code * 64;
    float s = 0.f;
#pragma unroll
    for (int c = 0; c < 64; ++c) s += e[c] * e[c];
    enorm[code] = s;
  } else if (tid < 65536 + 1024 + 128) {
    part[tid - 65536 - 1024] = 0.f;
  }
}

// ---------------------------------------------------------------------------
// Kernel 2: distances via v_wmma_f32_16x16x32_bf16, argmin, outputs.
// Two row-tiles per wave in flight -> 4 WMMAs per B-fragment load and
// two independent accumulator chains to cover WMMA hazard windows.
// ---------------------------------------------------------------------------
__launch_bounds__(256)
__global__ void vq_main(const float* __restrict__ z,
                        const float* __restrict__ emb,
                        char* __restrict__ ws,
                        float* __restrict__ out) {
  extern __shared__ char smem[];           // SMEM_BYTES: fragments + enorm
  __shared__ int   sIdx[WAVES_PER_BLOCK * 32];
  __shared__ float sWaveLoss[WAVES_PER_BLOCK];

  // Stage codebook fragments + norms into LDS with async global->LDS b128.
  // Dynamic LDS begins right after static LDS (groupstaticsize).
  {
    const unsigned dynBase = __builtin_amdgcn_groupstaticsize();
    for (int i = threadIdx.x; i < SMEM_BYTES / 16; i += 256)
      async_g2l_b128(dynBase + (unsigned)(i * 16), ws + (size_t)i * 16);
    wait_asynccnt0();
  }
  __syncthreads();

  const __bf16* fragS  = (const __bf16*)smem;
  const float*  enormS = (const float*)(smem + WS_ENORM);

  const int lane = threadIdx.x & 31;
  const int wave = threadIdx.x >> 5;
  const int h    = lane >> 4;              // lane half
  const int r    = lane & 15;              // row (A) / column (B,C,D) slot

  float* idxOut = out + N_ELEMS + 1;       // [N_ROWS] (loss scalar sits between)
  float lossAcc = 0.f;

  for (int jp = 0; jp < TILE_PAIRS; ++jp) {
    const int tile0 = ((blockIdx.x * WAVES_PER_BLOCK + wave) * TILE_PAIRS + jp) * 2;
    const int row0  = tile0 * 16;          // rows row0..row0+31 (never cross batch)
    const int b     = row0 >> 14;          // SPATIAL = 16384
    const int s     = row0 & (SPATIAL - 1);
    const float* zb = z + (size_t)b * C_DIM * SPATIAL + s;

    // A fragments for both tiles (tile A at +r, tile B at +16+r), bf16 in-register.
    v16bf a0, a1, c0, c1;
#pragma unroll
    for (int i = 0; i < 16; ++i) {
      const int k = i + 8 * h + ((i < 8) ? 0 : 8);      // chunk-0 K index
      a0[i] = f2bf(zb[(size_t)k        * SPATIAL + r]);
      a1[i] = f2bf(zb[(size_t)(k + 32) * SPATIAL + r]);
      c0[i] = f2bf(zb[(size_t)k        * SPATIAL + 16 + r]);
      c1[i] = f2bf(zb[(size_t)(k + 32) * SPATIAL + 16 + r]);
    }

    float bestSA[8], bestSB[8];
    int   bestIA[8], bestIB[8];
#pragma unroll
    for (int v = 0; v < 8; ++v) {
      bestSA[v] = 3.4e38f; bestIA[v] = 0;
      bestSB[v] = 3.4e38f; bestIB[v] = 0;
    }

    for (int ct = 0; ct < 64; ++ct) {
      const v16bf b0 = *(const v16bf*)(fragS + (ct * 2 + 0) * 512 + lane * 16);
      const v16bf b1 = *(const v16bf*)(fragS + (ct * 2 + 1) * 512 + lane * 16);
      v8f accA = {0.f, 0.f, 0.f, 0.f, 0.f, 0.f, 0.f, 0.f};
      v8f accB = {0.f, 0.f, 0.f, 0.f, 0.f, 0.f, 0.f, 0.f};
      accA = __builtin_amdgcn_wmma_f32_16x16x32_bf16(false, a0, false, b0,
                                                     (short)0, accA, false, false);
      accB = __builtin_amdgcn_wmma_f32_16x16x32_bf16(false, c0, false, b0,
                                                     (short)0, accB, false, false);
      accA = __builtin_amdgcn_wmma_f32_16x16x32_bf16(false, a1, false, b1,
                                                     (short)0, accA, false, false);
      accB = __builtin_amdgcn_wmma_f32_16x16x32_bf16(false, c1, false, b1,
                                                     (short)0, accB, false, false);
      const int   code = ct * 16 + r;      // this lane's column code id
      const float en   = enormS[code];
#pragma unroll
      for (int v = 0; v < 8; ++v) {
        const float sA = en - 2.0f * accA[v];   // ||e||^2 - 2 z.e (row const dropped)
        const float sB = en - 2.0f * accB[v];
        if (sA < bestSA[v]) { bestSA[v] = sA; bestIA[v] = code; }
        if (sB < bestSB[v]) { bestSB[v] = sB; bestIB[v] = code; }
      }
    }

    // Branchless tournament across the 16 lanes holding each row's columns.
#pragma unroll
    for (int v = 0; v < 8; ++v) {
      float sA = bestSA[v], sB = bestSB[v];
      int   iA = bestIA[v], iB = bestIB[v];
#pragma unroll
      for (int off = 8; off > 0; off >>= 1) {
        const float sA2 = __shfl_xor(sA, off, 32);
        const int   iA2 = __shfl_xor(iA, off, 32);
        const float sB2 = __shfl_xor(sB, off, 32);
        const int   iB2 = __shfl_xor(iB, off, 32);
        const bool tA = (sA2 < sA) | ((sA2 == sA) & (iA2 < iA));
        const bool tB = (sB2 < sB) | ((sB2 == sB) & (iB2 < iB));
        sA = tA ? sA2 : sA;  iA = tA ? iA2 : iA;
        sB = tB ? sB2 : sB;  iB = tB ? iB2 : iB;
      }
      bestIA[v] = iA;                       // uniform across the 16-lane half
      bestIB[v] = iB;
    }

    // Publish row indices (row m = h*8 + v) so every lane can gather emb[idx].
    if (r == 0) {
#pragma unroll
      for (int v = 0; v < 8; ++v) {
        sIdx[wave * 32 +      h * 8 + v] = bestIA[v];
        sIdx[wave * 32 + 16 + h * 8 + v] = bestIB[v];
      }
    }
    const int idxRowA = sIdx[wave * 32 + r];        // wave-local LDS RAW, in-order
    const int idxRowB = sIdx[wave * 32 + 16 + r];

    if (lane < 16) {
      idxOut[row0 + lane]      = (float)sIdx[wave * 32 + lane];
      idxOut[row0 + 16 + lane] = (float)sIdx[wave * 32 + 16 + lane];
    }

    // Quantized output (channel-first layout) + loss partial, both tiles.
    const float* eA = emb + (size_t)idxRowA * C_DIM;
    const float* eB = emb + (size_t)idxRowB * C_DIM;
    float*       zq = out + (size_t)b * C_DIM * SPATIAL + s;
#pragma unroll
    for (int cc = 0; cc < C_DIM; cc += 2) {
      const int c = cc + h;
      const float qA  = eA[c];
      const float zvA = zb[(size_t)c * SPATIAL + r];
      const float dA  = qA - zvA;
      lossAcc += dA * dA;
      zq[(size_t)c * SPATIAL + r] = qA;     // lanes cover 2 contiguous 64B runs
      const float qB  = eB[c];
      const float zvB = zb[(size_t)c * SPATIAL + 16 + r];
      const float dB  = qB - zvB;
      lossAcc += dB * dB;
      zq[(size_t)c * SPATIAL + 16 + r] = qB;
    }
  }

  // Deterministic loss reduction: wave shfl -> ordered per-block sum.
#pragma unroll
  for (int off = 16; off > 0; off >>= 1) lossAcc += __shfl_xor(lossAcc, off, 32);
  if (lane == 0) sWaveLoss[wave] = lossAcc;
  __syncthreads();
  if (threadIdx.x == 0) {
    float t = 0.f;
#pragma unroll
    for (int w = 0; w < WAVES_PER_BLOCK; ++w) t += sWaveLoss[w];
    ((float*)(ws + WS_PART))[blockIdx.x] = t;
  }
}

// ---------------------------------------------------------------------------
// Kernel 3: fold 128 partials into the scalar loss.
// ---------------------------------------------------------------------------
__global__ void vq_finish(const char* __restrict__ ws, float* __restrict__ out) {
  __shared__ float red[128];
  const float* part = (const float*)(ws + WS_PART);
  const int t = threadIdx.x;
  red[t] = part[t];
  __syncthreads();
  for (int off = 64; off > 0; off >>= 1) {
    if (t < off) red[t] += red[t + off];
    __syncthreads();
  }
  // vq_loss = codebook + 0.25*commitment = 1.25 * mean((zq - z)^2)
  if (t == 0) out[N_ELEMS] = red[0] * (1.25f / (float)N_ELEMS);
}

extern "C" void kernel_launch(void* const* d_in, const int* in_sizes, int n_in,
                              void* d_out, int out_size, void* d_ws, size_t ws_size,
                              hipStream_t stream) {
  (void)in_sizes; (void)n_in; (void)out_size; (void)ws_size;
  const float* z   = (const float*)d_in[0];   // (4,64,16,32,32) f32
  const float* emb = (const float*)d_in[1];   // (1024,64) f32
  float* out = (float*)d_out;                 // [zq | loss | indices]
  char*  ws  = (char*)d_ws;

  vq_prep<<<(65536 + 1024 + 128 + 255) / 256, 256, 0, stream>>>(emb, ws);
  vq_main<<<MAIN_BLOCKS, 256, SMEM_BYTES, stream>>>(z, emb, ws, out);
  vq_finish<<<1, 128, 0, stream>>>(ws, out);
}
